// WindowAttention_75093208203985
// MI455X (gfx1250) — matrix operand order repaired and checked by reading, not compiled, not run on hardware
//
#include <hip/hip_runtime.h>
#include <hip/hip_bf16.h>

typedef __bf16 bf16_t;
typedef __attribute__((ext_vector_type(16))) __bf16 v16bf;
typedef __attribute__((ext_vector_type(8)))  __bf16 v8bf;
typedef __attribute__((ext_vector_type(8)))  float  v8f;
typedef __attribute__((ext_vector_type(4)))  float  v4f;
typedef __attribute__((ext_vector_type(4)))  int    v4i;

#define CDIM 256

#if __has_builtin(__builtin_amdgcn_global_load_async_to_lds_b128) && \
    __has_builtin(__builtin_amdgcn_s_wait_asynccnt)
#define USE_ASYNC_LDS 1
typedef __attribute__((address_space(1))) v4i* as1_v4i_p;
typedef __attribute__((address_space(3))) v4i* as3_v4i_p;
#endif

static __device__ inline v8f zero8() {
  v8f z = {0.f, 0.f, 0.f, 0.f, 0.f, 0.f, 0.f, 0.f};
  return z;
}

// A fragment (16x32 bf16, M x K): lane l: M = l&15; K = {0..7,16..23} (l<16) or {8..15,24..31}
template<int LD>
static __device__ inline v16bf load_Afrag(const bf16_t* base, int lane) {
  const int half = lane >> 4, r = lane & 15;
  const bf16_t* p = base + r * LD + half * 8;
  v8bf lo = *(const v8bf*)(p);
  v8bf hi = *(const v8bf*)(p + 16);
  v16bf o;
#pragma unroll
  for (int i = 0; i < 8; ++i) { o[i] = lo[i]; o[8 + i] = hi[i]; }
  return o;
}

// B fragment (32x16 bf16, K x N), column N contiguous in K:
// lane l: N = l&15; K = {0..7,16..23} (l<16) or {8..15,24..31}
template<int LD>
static __device__ inline v16bf load_Bfrag(const bf16_t* base, int lane) {
  const int half = lane >> 4, c = lane & 15;
  const bf16_t* p = base + c * LD + half * 8;
  v8bf lo = *(const v8bf*)(p);
  v8bf hi = *(const v8bf*)(p + 16);
  v16bf o;
#pragma unroll
  for (int i = 0; i < 8; ++i) { o[i] = lo[i]; o[8 + i] = hi[i]; }
  return o;
}

static __device__ inline v8f wmma_bf16(v16bf a, v16bf b, v8f c) {
  return __builtin_amdgcn_wmma_f32_16x16x32_bf16(false, a, false, b, (short)0, c,
                                                 false, false);
}

static __device__ inline v8bf pack8(v4f a, v4f b) {
  v8bf p;
#pragma unroll
  for (int i = 0; i < 4; ++i) { p[i] = (bf16_t)a[i]; p[4 + i] = (bf16_t)b[i]; }
  return p;
}

// ---------------------------------------------------------------------------
// Kernel 1: fused QKV projection.  y = x @ [Wq;Wkv]^T + [bq;bkv]
// Grid (12, B): col tile fastest so 12 blocks sharing one x row-tile run
// back-to-back (x streams through HBM once; weights live in L2).
// q,k: [b][h][n][d] bf16.  v: TRANSPOSED [b][h][d][n] bf16 (for PV B-frags).
// ---------------------------------------------------------------------------
__global__ __launch_bounds__(256) void qkv_kernel(
    const float* __restrict__ x, const float* __restrict__ Wq,
    const float* __restrict__ bq, const float* __restrict__ Wkv,
    const float* __restrict__ bkv, bf16_t* __restrict__ qws,
    bf16_t* __restrict__ kws, bf16_t* __restrict__ vws) {
  __shared__ __align__(32) bf16_t xs[64 * 32];
  __shared__ __align__(32) bf16_t wt[64 * 32];
  const int tid = threadIdx.x, lane = tid & 31, wave = tid >> 5;
  const int it = wave & 3, jt0 = (wave >> 2) * 2;
  const int col0 = blockIdx.x * 64;  // output column in [0,768)
  const int row0 = blockIdx.y * 64;  // global token row = b*64 + n

  // Fixed staging slice per thread: row sr, 8 consecutive k at sc.
  const int sr = tid >> 2, sc = (tid & 3) * 8;
  const float* xrow = x + (size_t)(row0 + sr) * CDIM + sc;
  const int gc = col0 + sr;
  const float* wrow = (gc < CDIM) ? Wq + (size_t)gc * CDIM + sc
                                  : Wkv + (size_t)(gc - CDIM) * CDIM + sc;
  v8bf* xdst = (v8bf*)&xs[sr * 32 + sc];
  v8bf* wdst = (v8bf*)&wt[sr * 32 + sc];

  v8f acc0 = zero8(), acc1 = zero8();
  for (int kk = 0; kk < CDIM; kk += 32) {
    v4f x0 = *(const v4f*)(xrow + kk);
    v4f x1 = *(const v4f*)(xrow + kk + 4);
    v4f w0 = *(const v4f*)(wrow + kk);
    v4f w1 = *(const v4f*)(wrow + kk + 4);
    *xdst = pack8(x0, x1);
    *wdst = pack8(w0, w1);
    __syncthreads();
    v16bf a  = load_Afrag<32>(xs + it * 16 * 32, lane);
    v16bf b0 = load_Bfrag<32>(wt + (jt0 * 16) * 32, lane);
    v16bf b1 = load_Bfrag<32>(wt + ((jt0 + 1) * 16) * 32, lane);
    acc0 = wmma_bf16(a, b0, acc0);
    acc1 = wmma_bf16(a, b1, acc1);
    __syncthreads();
  }

  const int half = lane >> 4, ln = lane & 15;
#pragma unroll
  for (int t = 0; t < 2; ++t) {
    v8f acc = t ? acc1 : acc0;
    int col = col0 + (jt0 + t) * 16 + ln;
    float cb = (col < CDIM) ? bq[col] : bkv[col - CDIM];
#pragma unroll
    for (int j = 0; j < 8; ++j) {
      int row = row0 + it * 16 + half * 8 + j;
      int b = row >> 6, n = row & 63;
      float val = acc[j] + cb;
      if (col < CDIM) {
        int h = col >> 5, d = col & 31;
        qws[(((size_t)b * 8 + h) * 64 + n) * 32 + d] = (bf16_t)val;
      } else if (col < 2 * CDIM) {
        int cc = col - CDIM, h = cc >> 5, d = cc & 31;
        kws[(((size_t)b * 8 + h) * 64 + n) * 32 + d] = (bf16_t)val;
      } else {
        int cc = col - 2 * CDIM, h = cc >> 5, d = cc & 31;
        vws[(((size_t)b * 8 + h) * 32 + d) * 64 + n] = (bf16_t)val;  // transposed
      }
    }
  }
}

// Copy one 4KB head tile (2048 bf16) global -> LDS, 16B per lane per op.
static __device__ inline void tile_to_lds(const bf16_t* g, bf16_t* l, int lane) {
#ifdef USE_ASYNC_LDS
#pragma unroll
  for (int i = 0; i < 8; ++i) {
    __builtin_amdgcn_global_load_async_to_lds_b128(
        (as1_v4i_p)(g + (i * 32 + lane) * 8),
        (as3_v4i_p)(l + (i * 32 + lane) * 8), 0, 0);
  }
#else
#pragma unroll
  for (int i = 0; i < 8; ++i)
    *(v8bf*)(l + (i * 32 + lane) * 8) = *(const v8bf*)(g + (i * 32 + lane) * 8);
#endif
}

static __device__ inline void tile_wait() {
#ifdef USE_ASYNC_LDS
  __builtin_amdgcn_s_wait_asynccnt(0);
#endif
}

// ---------------------------------------------------------------------------
// Kernel 2: per-window attention. Grid B*2, 128 threads: one head per wave.
// q/k/v tiles are staged to LDS (async-to-LDS when available); P (bf16)
// overlays the q+k LDS region for the PV pass.
// ---------------------------------------------------------------------------
__global__ __launch_bounds__(128) void attn_kernel(
    const bf16_t* __restrict__ q, const bf16_t* __restrict__ k,
    const bf16_t* __restrict__ vt, const float* __restrict__ bias_table,
    const float* __restrict__ mask, bf16_t* __restrict__ ao, int nW) {
  __shared__ __align__(32) bf16_t smem[4][3 * 2048];  // q | k | v ; P overlays q|k
  const int b = blockIdx.x >> 1;
  const int wave = threadIdx.x >> 5;
  const int h = ((blockIdx.x & 1) << 2) + wave;
  const int lane = threadIdx.x & 31, half = lane >> 4, ln = lane & 15;
  bf16_t* qb = smem[wave];
  bf16_t* kb = qb + 2048;
  bf16_t* vb = kb + 2048;
  const size_t hoff = ((size_t)b * 8 + h) * 2048;
  const float* mrow = mask + (size_t)(b % nW) * 64 * 64;

  tile_to_lds(q + hoff, qb, lane);
  tile_to_lds(k + hoff, kb, lane);
  tile_to_lds(vt + hoff, vb, lane);
  tile_wait();

  // ---- S = q @ k^T  (16 WMMAs) ----
  v8f s[4][4];
  v16bf bfr[4];
#pragma unroll
  for (int jt = 0; jt < 4; ++jt) bfr[jt] = load_Bfrag<32>(kb + jt * 16 * 32, lane);
#pragma unroll
  for (int it = 0; it < 4; ++it) {
    v16bf a = load_Afrag<32>(qb + it * 16 * 32, lane);
#pragma unroll
    for (int jt = 0; jt < 4; ++jt) s[it][jt] = wmma_bf16(a, bfr[jt], zero8());
  }

  // ---- scale + relative-position bias + shift mask ----
  const float scale = 0.17677669529663687f;  // 32^-0.5
#pragma unroll
  for (int it = 0; it < 4; ++it)
#pragma unroll
    for (int jt = 0; jt < 4; ++jt)
#pragma unroll
      for (int j = 0; j < 8; ++j) {
        int r = it * 16 + half * 8 + j;
        int c = jt * 16 + ln;
        int idx = (((r >> 3) - (c >> 3) + 7) * 15) + ((r & 7) - (c & 7) + 7);
        s[it][jt][j] = s[it][jt][j] * scale + bias_table[idx * 8 + h] +
                       mrow[r * 64 + c];
      }

  // ---- row softmax: row spans lanes 0..15 (or 16..31) x 4 jt tiles ----
#pragma unroll
  for (int it = 0; it < 4; ++it)
#pragma unroll
    for (int j = 0; j < 8; ++j) {
      float m = fmaxf(fmaxf(s[it][0][j], s[it][1][j]),
                      fmaxf(s[it][2][j], s[it][3][j]));
#pragma unroll
      for (int off = 1; off < 16; off <<= 1) m = fmaxf(m, __shfl_xor(m, off, 32));
      float sum = 0.f;
#pragma unroll
      for (int jt = 0; jt < 4; ++jt) {
        float e = __expf(s[it][jt][j] - m);
        s[it][jt][j] = e;
        sum += e;
      }
#pragma unroll
      for (int off = 1; off < 16; off <<= 1) sum += __shfl_xor(sum, off, 32);
      float inv = 1.f / sum;
#pragma unroll
      for (int jt = 0; jt < 4; ++jt) s[it][jt][j] *= inv;
    }

  // ---- stage P (bf16) into q|k LDS region, A-fragment layout re-read ----
  bf16_t* pl = qb;  // 4096 elements: exactly q+k region
#pragma unroll
  for (int it = 0; it < 4; ++it)
#pragma unroll
    for (int jt = 0; jt < 4; ++jt)
#pragma unroll
      for (int j = 0; j < 8; ++j) {
        int r = it * 16 + half * 8 + j;
        int c = jt * 16 + ln;
        pl[r * 64 + c] = (bf16_t)s[it][jt][j];
      }

  // ---- O = P @ V  (16 WMMAs); V is [d][n] so B-frags are contiguous ----
  v8f ov[4][2];
#pragma unroll
  for (int it = 0; it < 4; ++it) { ov[it][0] = zero8(); ov[it][1] = zero8(); }
#pragma unroll
  for (int kc = 0; kc < 2; ++kc) {
    v16bf bv0 = load_Bfrag<64>(vb + (0 * 16) * 64 + kc * 32, lane);
    v16bf bv1 = load_Bfrag<64>(vb + (1 * 16) * 64 + kc * 32, lane);
#pragma unroll
    for (int it = 0; it < 4; ++it) {
      v16bf a = load_Afrag<64>(pl + it * 16 * 64 + kc * 32, lane);
      ov[it][0] = wmma_bf16(a, bv0, ov[it][0]);
      ov[it][1] = wmma_bf16(a, bv1, ov[it][1]);
    }
  }

  // ---- write [b][n][C], channel = h*32 + d ----
#pragma unroll
  for (int it = 0; it < 4; ++it)
#pragma unroll
    for (int dt = 0; dt < 2; ++dt)
#pragma unroll
      for (int j = 0; j < 8; ++j) {
        int n = it * 16 + half * 8 + j;
        int d = dt * 16 + ln;
        ao[(((size_t)b * 64 + n) * CDIM) + h * 32 + d] = (bf16_t)ov[it][dt][j];
      }
}

// ---------------------------------------------------------------------------
// Kernel 3: output projection. out = ao @ Wp^T + bp  (fp32 out)
// Grid (4, B): col tile fastest for ao reuse in L2.
// ---------------------------------------------------------------------------
__global__ __launch_bounds__(256) void proj_kernel(
    const bf16_t* __restrict__ ain, const float* __restrict__ Wp,
    const float* __restrict__ bp, float* __restrict__ out) {
  __shared__ __align__(32) bf16_t xs[64 * 32];
  __shared__ __align__(32) bf16_t wt[64 * 32];
  const int tid = threadIdx.x, lane = tid & 31, wave = tid >> 5;
  const int it = wave & 3, jt0 = (wave >> 2) * 2;
  const int col0 = blockIdx.x * 64;
  const int row0 = blockIdx.y * 64;

  const int sr = tid >> 2, sc = (tid & 3) * 8;
  const bf16_t* arow = ain + (size_t)(row0 + sr) * CDIM + sc;
  const float* wrow = Wp + (size_t)(col0 + sr) * CDIM + sc;
  v8bf* xdst = (v8bf*)&xs[sr * 32 + sc];
  v8bf* wdst = (v8bf*)&wt[sr * 32 + sc];

  v8f acc0 = zero8(), acc1 = zero8();
  for (int kk = 0; kk < CDIM; kk += 32) {
    v8bf av = *(const v8bf*)(arow + kk);
    v4f w0 = *(const v4f*)(wrow + kk);
    v4f w1 = *(const v4f*)(wrow + kk + 4);
    *xdst = av;
    *wdst = pack8(w0, w1);
    __syncthreads();
    v16bf a = load_Afrag<32>(xs + it * 16 * 32, lane);
    acc0 = wmma_bf16(a, load_Bfrag<32>(wt + jt0 * 16 * 32, lane), acc0);
    acc1 = wmma_bf16(a, load_Bfrag<32>(wt + (jt0 + 1) * 16 * 32, lane), acc1);
    __syncthreads();
  }

  const int half = lane >> 4, ln = lane & 15;
#pragma unroll
  for (int t = 0; t < 2; ++t) {
    v8f acc = t ? acc1 : acc0;
    int col = col0 + (jt0 + t) * 16 + ln;
    float cb = bp[col];
#pragma unroll
    for (int j = 0; j < 8; ++j) {
      int row = row0 + it * 16 + half * 8 + j;
      out[(size_t)row * CDIM + col] = acc[j] + cb;
    }
  }
}

extern "C" void kernel_launch(void* const* d_in, const int* in_sizes, int n_in,
                              void* d_out, int out_size, void* d_ws,
                              size_t ws_size, hipStream_t stream) {
  const float* x          = (const float*)d_in[0];
  const float* mask       = (const float*)d_in[1];
  const float* Wq         = (const float*)d_in[2];
  const float* bq         = (const float*)d_in[3];
  const float* Wkv        = (const float*)d_in[4];
  const float* bkv        = (const float*)d_in[5];
  const float* bias_table = (const float*)d_in[6];
  const float* Wp         = (const float*)d_in[7];
  const float* bp         = (const float*)d_in[8];
  float* out = (float*)d_out;

  const int B  = in_sizes[0] / (64 * 256);
  const int nW = in_sizes[1] / (64 * 64);

  const size_t per = (size_t)B * 8 * 64 * 32;  // bf16 elements per tensor
  bf16_t* qws  = (bf16_t*)d_ws;
  bf16_t* kws  = qws + per;
  bf16_t* vws  = kws + per;   // transposed [b][h][d][n]
  bf16_t* aows = vws + per;

  qkv_kernel<<<dim3(12, B), 256, 0, stream>>>(x, Wq, bq, Wkv, bkv, qws, kws, vws);
  attn_kernel<<<dim3(B * 2), 128, 0, stream>>>(qws, kws, vws, bias_table, mask,
                                               aows, nW);
  proj_kernel<<<dim3(4, B), 256, 0, stream>>>(aows, Wp, bp, out);
}
